// SequenceEmbedding_39505109189164
// MI455X (gfx1250) — matrix operand order repaired and checked by reading, not compiled, not run on hardware
//
#include <hip/hip_runtime.h>

// Problem dims (fixed by reference)
#define BATCH 16384
#define VDIM  8192   // K
#define DDIM  512    // N

// Tiling
#define BM 128
#define BN 128
#define BK 32
#define LDSS 40      // padded LDS row stride in halves (80B -> conflict-free)

typedef __attribute__((ext_vector_type(16))) _Float16 v16h;
typedef __attribute__((ext_vector_type(8)))  _Float16 v8h;
typedef __attribute__((ext_vector_type(8)))  float    v8f;
typedef __attribute__((ext_vector_type(4)))  int      i32x4;
typedef __attribute__((ext_vector_type(4)))  unsigned int u32x4;

union Frag { u32x4 q[2]; v16h h; };

__global__ __launch_bounds__(256)
void multihot_embed_wmma(const int* __restrict__ X,
                         const float* __restrict__ W,
                         float* __restrict__ Out)
{
    __shared__ __align__(16) _Float16 As[BM * LDSS];  // [m][k] f16 mask tile
    __shared__ __align__(16) _Float16 Bs[BN * LDSS];  // [n][k] transposed W tile (f16)

    const int tid  = threadIdx.x;
    const int lane = tid & 31;
    const int wave = tid >> 5;

    const int m0 = blockIdx.y * BM;
    const int n0 = blockIdx.x * BN;

    // 8 waves: 4 along M (32 rows each) x 2 along N (64 cols each)
    const int wm = (wave & 3) * 32;
    const int wn = (wave >> 2) * 64;

    // A-tile loader mapping: thread -> (row 0..127, k-half 0/16)
    const int a_row  = tid >> 1;
    const int a_half = (tid & 1) * 16;

    // B-tile loader mapping: thread -> (col 0..127, k-group 0/16)
    const int b_col = tid & 127;
    const int b_kg  = (tid >> 7) * 16;

    const int lrow = lane & 15;   // fragment row/col within 16
    const int lhi  = lane >> 4;   // lane half
    const int a_kb = lhi * 8;     // A frag K base (halves)
    const int b_kb = lhi * 8;     // B frag K base (halves)

    v8f zacc = {};
    v8f acc[2][4];
    #pragma unroll
    for (int i = 0; i < 2; ++i)
        #pragma unroll
        for (int j = 0; j < 4; ++j)
            acc[i][j] = zacc;

    for (int k0 = 0; k0 < VDIM; k0 += BK) {
        // ---- global loads (coalesced; in flight across the first barrier) ----
        const i32x4* xp =
            (const i32x4*)(X + (size_t)(m0 + a_row) * VDIM + (k0 + a_half));
        i32x4 xa = xp[0], xb = xp[1], xc = xp[2], xd = xp[3];

        float wv[16];
        const float* wp = W + (size_t)(k0 + b_kg) * DDIM + (n0 + b_col);
        #pragma unroll
        for (int i = 0; i < 16; ++i) wv[i] = wp[(size_t)i * DDIM];

        __syncthreads();  // previous iteration's LDS fragment reads complete

        // ---- convert + stage A (mask -> f16, exact) ----
        v8h pa0, pa1;
        #pragma unroll
        for (int i = 0; i < 4; ++i) {
            pa0[i]     = (xa[i] != 0) ? (_Float16)1.0f : (_Float16)0.0f;
            pa0[i + 4] = (xb[i] != 0) ? (_Float16)1.0f : (_Float16)0.0f;
            pa1[i]     = (xc[i] != 0) ? (_Float16)1.0f : (_Float16)0.0f;
            pa1[i + 4] = (xd[i] != 0) ? (_Float16)1.0f : (_Float16)0.0f;
        }
        *(v8h*)&As[a_row * LDSS + a_half]     = pa0;
        *(v8h*)&As[a_row * LDSS + a_half + 8] = pa1;

        // ---- convert + stage B transposed (f32 -> f16 RNE) ----
        v8h pb0, pb1;
        #pragma unroll
        for (int i = 0; i < 8; ++i) {
            pb0[i] = (_Float16)wv[i];
            pb1[i] = (_Float16)wv[i + 8];
        }
        *(v8h*)&Bs[b_col * LDSS + b_kg]     = pb0;
        *(v8h*)&Bs[b_col * LDSS + b_kg + 8] = pb1;

        __syncthreads();

        // ---- fragment loads (ISA 16-bit layouts) + 8 WMMAs ----
        Frag afrag[2];
        #pragma unroll
        for (int mf = 0; mf < 2; ++mf) {
            const _Float16* ap = &As[(wm + mf * 16 + lrow) * LDSS + a_kb];
            afrag[mf].q[0] = *(const u32x4*)ap;         // K = kb .. kb+7
            afrag[mf].q[1] = *(const u32x4*)(ap + 16);  // K = kb+16 .. kb+23
        }
        #pragma unroll
        for (int nf = 0; nf < 4; ++nf) {
            const _Float16* bp = &Bs[(wn + nf * 16 + lrow) * LDSS + b_kb];
            Frag bfrag;
            bfrag.q[0] = *(const u32x4*)bp;        // K = kb .. kb+7
            bfrag.q[1] = *(const u32x4*)(bp + 16); // K = kb+16 .. kb+23
            #pragma unroll
            for (int mf = 0; mf < 2; ++mf) {
                acc[mf][nf] = __builtin_amdgcn_wmma_f32_16x16x32_f16(
                    /*neg_a=*/false, afrag[mf].h,
                    /*neg_b=*/false, bfrag.h,
                    /*c_mod=*/(short)0, acc[mf][nf],
                    /*reuse_a=*/false, /*reuse_b=*/false);
            }
        }
    }

    // ---- epilogue: C/D layout is lanes 0-15 -> M=r, lanes 16-31 -> M=r+8 ----
    #pragma unroll
    for (int mf = 0; mf < 2; ++mf) {
        #pragma unroll
        for (int nf = 0; nf < 4; ++nf) {
            const int col = n0 + wn + nf * 16 + lrow;
            #pragma unroll
            for (int r = 0; r < 8; ++r) {
                const int row = m0 + wm + mf * 16 + lhi * 8 + r;
                Out[(size_t)row * DDIM + col] = acc[mf][nf][r];
            }
        }
    }
}

extern "C" void kernel_launch(void* const* d_in, const int* in_sizes, int n_in,
                              void* d_out, int out_size, void* d_ws, size_t ws_size,
                              hipStream_t stream) {
    (void)in_sizes; (void)n_in; (void)out_size; (void)d_ws; (void)ws_size;
    const int*   X = (const int*)d_in[0];   // [16384, 8192] int32 multi-hot
    const float* W = (const float*)d_in[1]; // [8192, 512] fp32 embeddings
    float*     Out = (float*)d_out;         // [16384, 512] fp32

    dim3 grid(DDIM / BN, BATCH / BM);       // (4, 128)
    multihot_embed_wmma<<<grid, 256, 0, stream>>>(X, W, Out);
}
